// LinearMoLoraLayer_35012573397735
// MI455X (gfx1250) — matrix operand order repaired and checked by reading, not compiled, not run on hardware
//
#include <hip/hip_runtime.h>

// ---------------------------------------------------------------------------
// MI455X (gfx1250) fused LinearMoLora:
//   out = x @ W_base^T + b_base + SCALING * ((x@A^T) * softmax(x@W_gate^T)) @ B
// Precision: f32 inputs split into bf16 hi/lo; products hi*hi + hi*lo + lo*hi
// accumulated in f32 via v_wmma_f32_16x16x32_bf16  (~f32 accuracy, bf16 rate).
// Pipeline: double-buffered LDS, 1 barrier / K-step, global prefetch into regs
// overlapped with the WMMA block of the previous step.
// ---------------------------------------------------------------------------

typedef __attribute__((ext_vector_type(16))) __bf16 v16bf;
typedef __attribute__((ext_vector_type(8)))  float  v8f;

#define M_TOK  8192      // 4 * 2048 tokens
#define D_DIM  4096      // d_in == d_out
#define E_EXP  8
#define R_RANK 16
#define KL     128       // E * R
#define MT     128       // M tile
#define NT     128       // N tile
#define KT     32        // K step (bf16 WMMA K)
#define LDAP   40        // padded LDS row stride (bf16 elems): 32 + 8
#define TILE_E (MT * LDAP)
#define SCALING 2.0f

union FragU { v16bf v; uint4 q[2]; };
union PackU { __bf16 h[16]; uint4 q[2]; };

// A-fragment (16x32 bf16, ISA layout): lane<16 holds M=lane, K={0..7,16..23};
// lane>=16 holds M=lane-16, K={8..15,24..31}.  LDS tile is row-major [row][K].
__device__ __forceinline__ v16bf load_frag_a(const __bf16* lds, int mBase, int lane) {
  const int row  = mBase + (lane & 15);
  const int half = (lane >> 4) << 3;            // 0 or 8
  FragU f;
  f.q[0] = *(const uint4*)(lds + row * LDAP + half);
  f.q[1] = *(const uint4*)(lds + row * LDAP + 16 + half);
  return f.v;
}

// B-fragment (32x16 bf16): lane<16 holds N=lane, K={0..7,16..23}; lane>=16 K={8..15,24..31}.
// Our LDS "W tile" is row-major [n][K], so this is two 16B chunks (same as A).
__device__ __forceinline__ v16bf load_frag_b(const __bf16* lds, int nBase, int lane) {
  const int row  = nBase + (lane & 15);
  const int half = (lane >> 4) << 3;            // 0 or 8
  FragU f;
  f.q[0] = *(const uint4*)(lds + row * LDAP + half);
  f.q[1] = *(const uint4*)(lds + row * LDAP + 16 + half);
  return f.v;
}

__device__ __forceinline__ void load16(const float* __restrict__ p, float* v) {
  const float4* p4 = (const float4*)p;
  float4 a = p4[0], b = p4[1], c = p4[2], d = p4[3];
  v[0]=a.x;  v[1]=a.y;  v[2]=a.z;  v[3]=a.w;
  v[4]=b.x;  v[5]=b.y;  v[6]=b.z;  v[7]=b.w;
  v[8]=c.x;  v[9]=c.y;  v[10]=c.z; v[11]=c.w;
  v[12]=d.x; v[13]=d.y; v[14]=d.z; v[15]=d.w;
}

// Split 16 f32 into bf16 hi/lo and store both 32B chunks to LDS.
__device__ __forceinline__ void split_store16(const float* v, __bf16* dhi, __bf16* dlo,
                                              int row, int kpart) {
  PackU uh, ul;
#pragma unroll
  for (int j = 0; j < 16; ++j) {
    float f = v[j];
    __bf16 hi = (__bf16)f;
    uh.h[j] = hi;
    ul.h[j] = (__bf16)(f - (float)hi);
  }
  __bf16* ph = dhi + row * LDAP + kpart;
  __bf16* pl = dlo + row * LDAP + kpart;
  *(uint4*)(ph)     = uh.q[0];
  *(uint4*)(ph + 8) = uh.q[1];
  *(uint4*)(pl)     = ul.q[0];
  *(uint4*)(pl + 8) = ul.q[1];
}

// hi*hi + hi*lo + lo*hi accumulated into f32 (lo*lo term below noise floor).
__device__ __forceinline__ v8f mm3(v16bf aH, v16bf aL, v16bf bH, v16bf bL, v8f c) {
  c = __builtin_amdgcn_wmma_f32_16x16x32_bf16(false, aH, false, bH, (short)0, c, false, false);
  c = __builtin_amdgcn_wmma_f32_16x16x32_bf16(false, aH, false, bL, (short)0, c, false, false);
  c = __builtin_amdgcn_wmma_f32_16x16x32_bf16(false, aL, false, bH, (short)0, c, false, false);
  return c;
}

// One full 128x128x32 tile-step for this wave: 12 frag loads, 24 WMMAs.
__device__ __forceinline__ void compute_step(const __bf16* sAhi, const __bf16* sAlo,
                                             const __bf16* sBhi, const __bf16* sBlo,
                                             int waveM, int waveN, int lane,
                                             v8f acc[4][2]) {
  v16bf aH[4], aL[4], bH[2], bL[2];
#pragma unroll
  for (int mf = 0; mf < 4; ++mf) {
    aH[mf] = load_frag_a(sAhi, waveM * 64 + mf * 16, lane);
    aL[mf] = load_frag_a(sAlo, waveM * 64 + mf * 16, lane);
  }
#pragma unroll
  for (int nf = 0; nf < 2; ++nf) {
    bH[nf] = load_frag_b(sBhi, waveN * 32 + nf * 16, lane);
    bL[nf] = load_frag_b(sBlo, waveN * 32 + nf * 16, lane);
  }
#pragma unroll
  for (int mf = 0; mf < 4; ++mf)
#pragma unroll
    for (int nf = 0; nf < 2; ++nf)
      acc[mf][nf] = mm3(aH[mf], aL[mf], bH[nf], bL[nf], acc[mf][nf]);
}

// ---------------------------------------------------------------------------
// Kernel 1: gating.  route[m][e] = softmax_e( dot(x[m], W_gate[e]) )
// 8 tokens per 256-thread block (1 wave per token), ds_add_f32 reduction.
// ---------------------------------------------------------------------------
__global__ __launch_bounds__(256) void gate_softmax(const float* __restrict__ x,
                                                    const float* __restrict__ Wg,
                                                    float* __restrict__ route) {
  __shared__ float sRed[8][8];
  const int t = threadIdx.x, lane = t & 31, tok = t >> 5;
  const size_t m = (size_t)blockIdx.x * 8 + tok;

  if (t < 64) ((float*)sRed)[t] = 0.0f;
  __syncthreads();

  float acc[E_EXP] = {};
  const float4* xp = (const float4*)(x + m * D_DIM);
  const float4* gp = (const float4*)Wg;
#pragma unroll 4
  for (int i = 0; i < 32; ++i) {
    const int k4 = lane + 32 * i;              // float4 index, covers K=4096
    float4 xv = xp[k4];
#pragma unroll
    for (int e = 0; e < E_EXP; ++e) {
      float4 wv = gp[e * (D_DIM / 4) + k4];
      acc[e] += xv.x * wv.x + xv.y * wv.y + xv.z * wv.z + xv.w * wv.w;
    }
  }
#pragma unroll
  for (int e = 0; e < E_EXP; ++e) atomicAdd(&sRed[tok][e], acc[e]);
  __syncthreads();

  if (t < 8) {
    float v[E_EXP], mx = -3.0e38f;
#pragma unroll
    for (int e = 0; e < E_EXP; ++e) { v[e] = sRed[t][e]; mx = fmaxf(mx, v[e]); }
    float s = 0.0f;
#pragma unroll
    for (int e = 0; e < E_EXP; ++e) { v[e] = __expf(v[e] - mx); s += v[e]; }
    const float inv = 1.0f / s;
    const size_t mm = (size_t)blockIdx.x * 8 + t;
#pragma unroll
    for (int e = 0; e < E_EXP; ++e) route[mm * E_EXP + e] = v[e] * inv;
  }
}

// ---------------------------------------------------------------------------
// Kernel 2: LoRA down.  hw[m][e*16+r] = SCALING*route[m][e]*dot(x[m], A[e][r])
// Double-buffered WMMA GEMM, M-tiles of 128, N = 128 (E*R), K = 4096.
// ---------------------------------------------------------------------------
__global__ __launch_bounds__(256) void lora_down(const float* __restrict__ x,
                                                 const float* __restrict__ Amat,
                                                 const float* __restrict__ route,
                                                 float* __restrict__ hw) {
  __shared__ __bf16 sAhi[2 * TILE_E];
  __shared__ __bf16 sAlo[2 * TILE_E];
  __shared__ __bf16 sBhi[2 * TILE_E];
  __shared__ __bf16 sBlo[2 * TILE_E];

  const int t = threadIdx.x;
  const int lane = t & 31, wave = t >> 5;
  const int waveM = wave >> 2, waveN = wave & 3;
  const size_t m0 = (size_t)blockIdx.x * MT;
  const int srow  = t >> 1;
  const int kpart = (t & 1) << 4;

  v8f acc[4][2] = {};
  const int KS = D_DIM / KT;                    // 128

  // prologue: stage step 0 into buffer 0
  {
    float va[16], vb[16];
    load16(x    + (m0 + srow) * D_DIM + kpart, va);
    load16(Amat + (size_t)srow * D_DIM + kpart, vb);
    split_store16(va, sAhi, sAlo, srow, kpart);
    split_store16(vb, sBhi, sBlo, srow, kpart);
  }

  for (int ks = 0; ks < KS; ++ks) {
    __syncthreads();                            // buffer (ks&1) now visible
    const int cur = (ks & 1) * TILE_E;
    const int nxt = ((ks + 1) & 1) * TILE_E;

    float na[16], nb[16];
    const bool more = (ks + 1) < KS;
    if (more) {                                 // issue next-step globals early
      const int k0 = (ks + 1) * KT + kpart;
      const float* pa = x    + (m0 + srow) * D_DIM + k0;
      const float* pb = Amat + (size_t)srow * D_DIM + k0;
      load16(pa, na);
      load16(pb, nb);
      __builtin_prefetch(pa + KT, 0, 1);
    }

    compute_step(sAhi + cur, sAlo + cur, sBhi + cur, sBlo + cur,
                 waveM, waveN, lane, acc);      // hides global latency

    if (more) {
      split_store16(na, sAhi + nxt, sAlo + nxt, srow, kpart);
      split_store16(nb, sBhi + nxt, sBlo + nxt, srow, kpart);
    }
  }

  // epilogue: fold SCALING * route into the low-rank intermediate
#pragma unroll
  for (int nf = 0; nf < 2; ++nf) {
    const int col = waveN * 32 + nf * 16 + (lane & 15);
    const int e   = col >> 4;
#pragma unroll
    for (int mf = 0; mf < 4; ++mf) {
#pragma unroll
      for (int j = 0; j < 8; ++j) {
        const size_t r = m0 + waveM * 64 + mf * 16 + ((lane >> 4) << 3) + j;
        hw[r * KL + col] = SCALING * route[r * E_EXP + e] * acc[mf][nf][j];
      }
    }
  }
}

// ---------------------------------------------------------------------------
// Kernel 3: fused base GEMM + LoRA up as a K-extended GEMM:
//   out = [x | hw] @ [W_base^T | B_perm] + b_base,  K = 4096 + 128
// where B_perm[o][e*16+r] = B[e][o][r] (contiguous 16-float gather per row).
// ---------------------------------------------------------------------------
__device__ __forceinline__ void main_ptrs(int ks, size_t m0, size_t n0, int srow, int kpart,
                                          const float* __restrict__ x,
                                          const float* __restrict__ Wb,
                                          const float* __restrict__ hw,
                                          const float* __restrict__ Bm,
                                          const float** pa, const float** pb) {
  const int KS_BASE = D_DIM / KT;
  if (ks < KS_BASE) {
    const int k0 = ks * KT + kpart;
    *pa = x  + (m0 + srow) * D_DIM + k0;
    *pb = Wb + (n0 + srow) * D_DIM + k0;
  } else {
    const int kk = (ks - KS_BASE) * KT + kpart;   // 0..112, 16-aligned
    const int e  = kk >> 4;                       // expert id
    *pa = hw + (m0 + srow) * KL + kk;             // route-scaled intermediate
    *pb = Bm + ((size_t)e * D_DIM + (n0 + srow)) * R_RANK;  // B[e][n][0..15]
  }
}

__global__ __launch_bounds__(256) void fused_base_lora(const float* __restrict__ x,
                                                       const float* __restrict__ Wb,
                                                       const float* __restrict__ bb,
                                                       const float* __restrict__ Bm,
                                                       const float* __restrict__ hw,
                                                       float* __restrict__ out) {
  __shared__ __bf16 sAhi[2 * TILE_E];
  __shared__ __bf16 sAlo[2 * TILE_E];
  __shared__ __bf16 sBhi[2 * TILE_E];
  __shared__ __bf16 sBlo[2 * TILE_E];

  const int t = threadIdx.x;
  const int lane = t & 31, wave = t >> 5;
  const int waveM = wave >> 2, waveN = wave & 3;
  const size_t m0 = (size_t)blockIdx.y * MT;
  const size_t n0 = (size_t)blockIdx.x * NT;
  const int srow  = t >> 1;
  const int kpart = (t & 1) << 4;

  v8f acc[4][2] = {};
  const int KS = D_DIM / KT + KL / KT;          // 132

  // prologue: stage step 0 into buffer 0
  {
    const float *pa, *pb;
    main_ptrs(0, m0, n0, srow, kpart, x, Wb, hw, Bm, &pa, &pb);
    float va[16], vb[16];
    load16(pa, va);
    load16(pb, vb);
    split_store16(va, sAhi, sAlo, srow, kpart);
    split_store16(vb, sBhi, sBlo, srow, kpart);
  }

  for (int ks = 0; ks < KS; ++ks) {
    __syncthreads();                            // buffer (ks&1) now visible
    const int cur = (ks & 1) * TILE_E;
    const int nxt = ((ks + 1) & 1) * TILE_E;

    float na[16], nb[16];
    const bool more = (ks + 1) < KS;
    if (more) {                                 // issue next-step globals early
      const float *pa, *pb;
      main_ptrs(ks + 1, m0, n0, srow, kpart, x, Wb, hw, Bm, &pa, &pb);
      load16(pa, na);
      load16(pb, nb);
      __builtin_prefetch(pa + KT, 0, 1);
      __builtin_prefetch(pb + KT, 0, 1);
    }

    compute_step(sAhi + cur, sAlo + cur, sBhi + cur, sBlo + cur,
                 waveM, waveN, lane, acc);      // hides global latency

    if (more) {
      split_store16(na, sAhi + nxt, sAlo + nxt, srow, kpart);
      split_store16(nb, sBhi + nxt, sBlo + nxt, srow, kpart);
    }
  }

  // epilogue: bias + store f32
#pragma unroll
  for (int nf = 0; nf < 2; ++nf) {
    const size_t c = n0 + waveN * 32 + nf * 16 + (lane & 15);
    const float bias = bb[c];
#pragma unroll
    for (int mf = 0; mf < 4; ++mf) {
#pragma unroll
      for (int j = 0; j < 8; ++j) {
        const size_t r = m0 + waveM * 64 + mf * 16 + ((lane >> 4) << 3) + j;
        out[r * D_DIM + c] = acc[mf][nf][j] + bias;
      }
    }
  }
}

// ---------------------------------------------------------------------------
extern "C" void kernel_launch(void* const* d_in, const int* in_sizes, int n_in,
                              void* d_out, int out_size, void* d_ws, size_t ws_size,
                              hipStream_t stream) {
  const float* x  = (const float*)d_in[0];   // [4,2048,4096]
  const float* Wb = (const float*)d_in[1];   // [4096,4096]
  const float* bb = (const float*)d_in[2];   // [4096]
  const float* Wg = (const float*)d_in[3];   // [8,4096]
  const float* A  = (const float*)d_in[4];   // [8,16,4096] == [128,4096]
  const float* Bm = (const float*)d_in[5];   // [8,4096,16]
  float* out = (float*)d_out;                // [4,2048,4096]

  float* route = (float*)d_ws;               // [8192,8]
  float* hw    = route + (size_t)M_TOK * E_EXP;  // [8192,128]

  gate_softmax<<<M_TOK / 8, 256, 0, stream>>>(x, Wg, route);
  lora_down<<<M_TOK / MT, 256, 0, stream>>>(x, A, route, hw);
  fused_base_lora<<<dim3(D_DIM / NT, M_TOK / MT), 256, 0, stream>>>(x, Wb, bb, Bm, hw, out);
}
